// LSTMModel_45466523795553
// MI455X (gfx1250) — matrix operand order repaired and checked by reading, not compiled, not run on hardware
//
#include <hip/hip_runtime.h>
#include <hip/hip_bf16.h>

typedef __attribute__((ext_vector_type(16))) _Float16 v16h;
typedef __attribute__((ext_vector_type(8)))  _Float16 v8h;
typedef __attribute__((ext_vector_type(8)))  float    v8f;
typedef __attribute__((ext_vector_type(4)))  float    v4f;

#define HID 64
#define GATES 256          // 4*HID
#define TSTEPS 10
#define WAVES_PER_BLOCK 8
#define ROWS_PER_WAVE 16
#define ROWS_PER_BLOCK (WAVES_PER_BLOCK * ROWS_PER_WAVE)

// Activations on the recurrence's critical path: prefer the native CDNA5
// V_TANH transcendental if the toolchain exposes it; otherwise v_exp+v_rcp.
#if __has_builtin(__builtin_amdgcn_tanhf)
__device__ __forceinline__ float fast_tanh(float x) {
    return __builtin_amdgcn_tanhf(x);                 // v_tanh_f32
}
__device__ __forceinline__ float fast_sigmoid(float x) {
    return __builtin_fmaf(0.5f, __builtin_amdgcn_tanhf(0.5f * x), 0.5f);
}
#else
__device__ __forceinline__ float fast_sigmoid(float x) {
    return __builtin_amdgcn_rcpf(1.0f + __expf(-x));
}
__device__ __forceinline__ float fast_tanh(float x) {
    return 2.0f * __builtin_amdgcn_rcpf(1.0f + __expf(-2.0f * x)) - 1.0f;
}
#endif

// B-fragment (32x16 f16) for tile with gate-column base n0, K-chunk kk, from
// row-major [256][64] f16 weights in LDS. Per ISA: lane holds N = lane%16,
// K = kk*32 + (j/8)*16 + (lane/16)*8 + (j%8) for j=0..15  -> two 16B LDS reads.
__device__ __forceinline__ v16h load_bfrag(const _Float16* W, int n0, int kk, int lane) {
    int n = n0 + (lane & 15);
    int kbase = kk * 32 + ((lane >> 4) << 3);
    v8h a = *(const v8h*)(W + n * HID + kbase);
    v8h b = *(const v8h*)(W + n * HID + kbase + 16);
    v16h r;
#pragma unroll
    for (int j = 0; j < 8; ++j) { r[j] = a[j]; r[8 + j] = b[j]; }
    return r;
}

// A-fragment (16x32 f16) straight from a global fp32 row (one batch row of x).
// Per ISA 16-bit A layout: lane (group g = lane/16) element j maps to
// K = kk*32 + (j/8)*16 + g*8 + (j%8)  -> two contiguous 8-float runs.
__device__ __forceinline__ v16h load_afrag_global_f32(const float* row, int kk, int g) {
    const float* p0 = row + kk * 32 + g * 8;
    const float* p1 = row + kk * 32 + 16 + g * 8;
    v4f a0 = *(const v4f*)(p0);
    v4f a1 = *(const v4f*)(p0 + 4);
    v4f b0 = *(const v4f*)(p1);
    v4f b1 = *(const v4f*)(p1 + 4);
    v16h r;
#pragma unroll
    for (int j = 0; j < 4; ++j) {
        r[j]      = (_Float16)a0[j];
        r[4 + j]  = (_Float16)a1[j];
        r[8 + j]  = (_Float16)b0[j];
        r[12 + j] = (_Float16)b1[j];
    }
    return r;
}

// Same A-fragment pattern, but from the per-wave row-major [16][64] f16 h tile in LDS.
__device__ __forceinline__ v16h load_afrag_lds_f16(const _Float16* hlds, int kk, int lane) {
    int m = lane & 15;
    int g = lane >> 4;
    const _Float16* p0 = hlds + m * HID + kk * 32 + g * 8;
    v8h a = *(const v8h*)(p0);
    v8h b = *(const v8h*)(p0 + 16);
    v16h r;
#pragma unroll
    for (int j = 0; j < 8; ++j) { r[j] = a[j]; r[8 + j] = b[j]; }
    return r;
}

__global__ __launch_bounds__(256) void lstm_wmma_kernel(
    const float* __restrict__ x, const float* __restrict__ W_ih,
    const float* __restrict__ W_hh, const float* __restrict__ b_ih,
    const float* __restrict__ b_hh, const float* __restrict__ W_out,
    const float* __restrict__ b_out, float* __restrict__ out, int B)
{
    __shared__ _Float16 Wih[GATES * HID];                            // 32 KB
    __shared__ _Float16 Whh[GATES * HID];                            // 32 KB
    __shared__ _Float16 hall[WAVES_PER_BLOCK * ROWS_PER_WAVE * HID]; // 16 KB

    const int tid  = threadIdx.x;
    const int wave = tid >> 5;
    const int lane = tid & 31;

    // Stage weights fp32 -> f16 into LDS: thread t converts row t of each matrix.
    {
        const float* sih = W_ih + tid * HID;
        const float* shh = W_hh + tid * HID;
        _Float16* dih = Wih + tid * HID;
        _Float16* dhh = Whh + tid * HID;
#pragma unroll
        for (int j = 0; j < HID; j += 4) {
            v4f a = *(const v4f*)(sih + j);
            v4f b = *(const v4f*)(shh + j);
#pragma unroll
            for (int q = 0; q < 4; ++q) {
                dih[j + q] = (_Float16)a[q];
                dhh[j + q] = (_Float16)b[q];
            }
        }
    }
    __syncthreads();

    const int batch0 = (blockIdx.x * WAVES_PER_BLOCK + wave) * ROWS_PER_WAVE;
    if (batch0 + ROWS_PER_WAVE > B) return;   // wave-uniform: EXEC stays all-ones for WMMA

    _Float16* hlds = hall + wave * (ROWS_PER_WAVE * HID);

    // h0 = 0
    {
        v8h z = {};
#pragma unroll
        for (int i = 0; i < 4; ++i) *(((v8h*)hlds) + lane * 4 + i) = z;
    }

    // Per-lane bias registers: C/D fragment has N = n0 + lane%16 for all 8 rows.
    float biasr[16];
#pragma unroll
    for (int gate = 0; gate < 4; ++gate)
#pragma unroll
        for (int hc = 0; hc < 4; ++hc) {
            int n = gate * 64 + hc * 16 + (lane & 15);
            biasr[gate * 4 + hc] = b_ih[n] + b_hh[n];
        }

    const float* xlane = x + (size_t)(batch0 + (lane & 15)) * (TSTEPS * HID);
    const int g = lane >> 4;

    v8f c_state[4];
#pragma unroll
    for (int hc = 0; hc < 4; ++hc) c_state[hc] = (v8f){0, 0, 0, 0, 0, 0, 0, 0};
    v8f h_frag[4];

    for (int t = 0; t < TSTEPS; ++t) {
        // Kill LICM of the 64 loop-invariant weight fragments: without this the
        // compiler hoists 512 VGPRs of B-fragments and spills them to scratch.
        // With it, fragments are re-read from LDS each step (cheap, high reuse).
        asm volatile("" ::: "memory");

        const float* xrow = xlane + t * HID;
        if (t + 1 < TSTEPS) {
            __builtin_prefetch(xrow + HID, 0, 1);            // global_prefetch_b8
            __builtin_prefetch(xrow + HID + 32, 0, 1);
        }

        v16h xa0 = load_afrag_global_f32(xrow, 0, g);
        v16h xa1 = load_afrag_global_f32(xrow, 1, g);
        v16h ha0 = load_afrag_lds_f16(hlds, 0, lane);   // reads ALL of h_t before
        v16h ha1 = load_afrag_lds_f16(hlds, 1, lane);   // h_{t+1} is written below

#pragma unroll
        for (int hc = 0; hc < 4; ++hc) {
            v8f z[4];
#pragma unroll
            for (int gate = 0; gate < 4; ++gate) {
                int n0 = gate * 64 + hc * 16;
                float bv = biasr[gate * 4 + hc];
                v8f acc = { bv, bv, bv, bv, bv, bv, bv, bv };
                acc = __builtin_amdgcn_wmma_f32_16x16x32_f16(
                    false, xa0, false, load_bfrag(Wih, n0, 0, lane), (short)0, acc, false, false);
                acc = __builtin_amdgcn_wmma_f32_16x16x32_f16(
                    false, xa1, false, load_bfrag(Wih, n0, 1, lane), (short)0, acc, false, false);
                acc = __builtin_amdgcn_wmma_f32_16x16x32_f16(
                    false, ha0, false, load_bfrag(Whh, n0, 0, lane), (short)0, acc, false, false);
                acc = __builtin_amdgcn_wmma_f32_16x16x32_f16(
                    false, ha1, false, load_bfrag(Whh, n0, 1, lane), (short)0, acc, false, false);
                z[gate] = acc;
            }
            v8f cs = c_state[hc];
            v8f hh;
#pragma unroll
            for (int r = 0; r < 8; ++r) {
                float ig = fast_sigmoid(z[0][r]);
                float fg = fast_sigmoid(z[1][r]);
                float gg = fast_tanh(z[2][r]);
                float og = fast_sigmoid(z[3][r]);
                float c  = fg * cs[r] + ig * gg;
                cs[r] = c;
                hh[r] = og * fast_tanh(c);
            }
            c_state[hc] = cs;
            h_frag[hc]  = hh;
        }

        // Transpose h fragment (C/D layout) back to row-major f16 in LDS for the
        // next step's A-fragment: lane element r lives at (m = r+8g, n = hc*16+lane%16).
#pragma unroll
        for (int hc = 0; hc < 4; ++hc)
#pragma unroll
            for (int r = 0; r < 8; ++r) {
                int m = r + 8 * g;
                int n = hc * 16 + (lane & 15);
                hlds[m * HID + n] = (_Float16)h_frag[hc][r];
            }
    }

    // Output projection: lanes 0..15 each finish one batch row.
    if (lane < 16) {
        float sum = b_out[0];
#pragma unroll
        for (int k = 0; k < HID; ++k) sum += (float)hlds[lane * HID + k] * W_out[k];
        out[batch0 + lane] = sum;
    }
}

extern "C" void kernel_launch(void* const* d_in, const int* in_sizes, int n_in,
                              void* d_out, int out_size, void* d_ws, size_t ws_size,
                              hipStream_t stream) {
    const float* x     = (const float*)d_in[0];
    const float* W_ih  = (const float*)d_in[1];
    const float* W_hh  = (const float*)d_in[2];
    const float* b_ih  = (const float*)d_in[3];
    const float* b_hh  = (const float*)d_in[4];
    const float* W_out = (const float*)d_in[5];
    const float* b_out = (const float*)d_in[6];
    float* out = (float*)d_out;

    int B = in_sizes[0] / (TSTEPS * HID);                    // 131072
    int blocks = (B + ROWS_PER_BLOCK - 1) / ROWS_PER_BLOCK;  // 1024
    lstm_wmma_kernel<<<blocks, 256, 0, stream>>>(x, W_ih, W_hh, b_ih, b_hh,
                                                 W_out, b_out, out, B);
}
